// Attention_43173011259925
// MI455X (gfx1250) — matrix operand order repaired and checked by reading, not compiled, not run on hardware
//
#include <hip/hip_runtime.h>
#include <stdint.h>
#include <math.h>

#define B_ 32
#define L_ 2048
#define E_ 512
#define F_ 1024          // E + Q
#define MT 64            // rows per workgroup in the scores kernel
#define AS 1040          // padded LDS row stride in bf16 elements (520 dwords; %64 == 8)

typedef __attribute__((ext_vector_type(4)))  unsigned int u32x4;
typedef __attribute__((ext_vector_type(2)))  unsigned int u32x2;
typedef __attribute__((ext_vector_type(4)))  float        f32x4;
typedef __attribute__((ext_vector_type(8)))  float        v8f;
typedef __attribute__((ext_vector_type(16))) __bf16       v16bf;
typedef __attribute__((ext_vector_type(4)))  unsigned int v4u;
typedef __attribute__((ext_vector_type(8)))  int          v8i;
typedef __attribute__((ext_vector_type(4)))  int          v4i;

// hardware tanh if this toolchain exposes it (gfx1250 has a TANH trans op)
#if __has_builtin(__builtin_amdgcn_tanhf)
#define FAST_TANH(x) __builtin_amdgcn_tanhf(x)
#else
#define FAST_TANH(x) tanhf(x)
#endif

// round-to-nearest-even f32 -> bf16, packed pair
static __device__ __forceinline__ unsigned pack_bf2(float lo, float hi) {
  unsigned a = __float_as_uint(lo), b = __float_as_uint(hi);
  a += 0x7FFFu + ((a >> 16) & 1u);
  b += 0x7FFFu + ((b >> 16) & 1u);
  return (a >> 16) | (b & 0xFFFF0000u);
}

// ---------------------------------------------------------------------------
// Kernel 0: convert W_attn (512 x 1024 f32) -> bf16 in workspace
// ---------------------------------------------------------------------------
__global__ void __launch_bounds__(256)
wprep_kernel(const float* __restrict__ W, unsigned* __restrict__ Wbf) {
  int i = blockIdx.x * blockDim.x + threadIdx.x;     // 131072 float4 chunks
  f32x4 w = *(const f32x4*)(W + (size_t)i * 4);
  u32x2 o;
  o.x = pack_bf2(w.x, w.y);
  o.y = pack_bf2(w.z, w.w);
  *(u32x2*)(Wbf + (size_t)i * 2) = o;
}

// ---------------------------------------------------------------------------
// TDM: async-load one 16x1024 bf16 W tile into LDS with padding so the LDS
// row stride becomes 1040 bf16 (2048B data + 2x16B pads) -> conflict-optimal
// ds_load_b128 WMMA-fragment reads.
// ---------------------------------------------------------------------------
static __device__ __forceinline__ void tdm_issue_btile(const unsigned short* gsrc,
                                                       const unsigned short* lds_dst) {
  unsigned long long ga = (unsigned long long)(uintptr_t)gsrc;
  unsigned la = (unsigned)(uintptr_t)lds_dst;  // low 32 bits of shared addr == LDS byte offset
  v4u g0;
  g0.x = 1u;                                                   // count=1, user descriptor
  g0.y = la;                                                   // lds_addr
  g0.z = (unsigned)ga;                                         // global_addr[31:0]
  g0.w = ((unsigned)(ga >> 32) & 0x01FFFFFFu) | 0x80000000u;   // addr[56:32] | type=2
  v8i g1;
  g1[0] = (int)((1u << 16) |   // data_size = 2 bytes
                (1u << 20) |   // pad_enable
                (7u << 22) |   // pad_interval: every 256 dwords (1024 B)
                (3u << 25));   // pad_amount:  4 dwords (16 B)
  g1[1] = (int)(1024u << 16);  // tensor_dim0 = 1024 (low16)
  g1[2] = (int)(16u << 16);    // tensor_dim1 = 16   (low16)
  g1[3] = (int)(1024u << 16);  // tile_dim0   = 1024
  g1[4] = 16;                  // tile_dim1   = 16, tile_dim2 = 0
  g1[5] = 1024;                // tensor_dim0_stride = 1024 elements
  g1[6] = 0;
  g1[7] = 0;
  v4i z4 = {0, 0, 0, 0};
  v8i z8 = {0, 0, 0, 0, 0, 0, 0, 0};
  __builtin_amdgcn_tensor_load_to_lds(g0, g1, z4, z4, z8, 0);
}

// ---------------------------------------------------------------------------
// Kernel 1: scores[m] = sum_n tanh( (cat @ W^T)[m,n] + b[n] ) * v[n]
// 64 rows per workgroup, full K=1024 A-tile resident in LDS as bf16,
// W tiles streamed by TDM (double-buffered per n-stream), WMMA bf16 core.
// ---------------------------------------------------------------------------
__global__ void __launch_bounds__(256)
attn_scores_kernel(const float* __restrict__ qry, const float* __restrict__ emb,
                   const unsigned short* __restrict__ Wbf,
                   const float* __restrict__ bias, const float* __restrict__ vwv,
                   float* __restrict__ scores) {
  __shared__ unsigned short Ash[MT * AS];        // 133,120 B
  __shared__ unsigned short Bsh[4][16 * AS];     // 133,120 B (2 streams x 2 buffers)
  __shared__ float sBias[E_], sV[E_], sScore[MT];

  const int tid     = threadIdx.x;
  const int wave    = tid >> 5;
  const int lane    = tid & 31;
  const int rowBase = blockIdx.x * MT;
  const int stream  = wave >> 2;   // 0: n-tiles 0..15 ; 1: n-tiles 16..31
  const int msub    = wave & 3;    // 16-row M subtile

  // kick off first TDM B-tile loads; they stream while we stage/convert A
  if (wave == 0) tdm_issue_btile(Wbf + (size_t)0 * 16 * F_,  &Bsh[0][0]);
  if (wave == 4) tdm_issue_btile(Wbf + (size_t)16 * 16 * F_, &Bsh[2][0]);

  // stage A tile: 64 rows x 1024 (query | embeddings), f32 -> bf16 into LDS
  for (int i = tid; i < MT * (F_ / 4); i += 256) {  // 16384 float4 chunks
    int row = i >> 8;            // 256 float4 per row
    int col = (i & 255) * 4;
    const float* src = (col < 512) ? (qry + (size_t)(rowBase + row) * 512 + col)
                                   : (emb + (size_t)(rowBase + row) * 512 + (col - 512));
    f32x4 v = *(const f32x4*)src;
    u32x2 p;
    p.x = pack_bf2(v.x, v.y);
    p.y = pack_bf2(v.z, v.w);
    *(u32x2*)&Ash[row * AS + col] = p;
  }
  for (int i = tid; i < E_; i += 256) { sBias[i] = bias[i]; sV[i] = vwv[i]; }
  if (tid < MT) sScore[tid] = 0.f;
  __syncthreads();

  const int g    = lane >> 4;          // K-half group within the wave
  const int nrow = lane & 15;          // B column within tile / C column
  const unsigned short* aBase = &Ash[(msub * 16 + (lane & 15)) * AS + g * 8];

  float sp[8];
#pragma unroll
  for (int r = 0; r < 8; ++r) sp[r] = 0.f;

  for (int t = 0; t < 16; ++t) {
    // prefetch next tile of this stream, then wait until tile t has landed
    if (wave == 0) {
      if (t + 1 < 16) {
        tdm_issue_btile(Wbf + (size_t)(t + 1) * (16 * F_), &Bsh[(t + 1) & 1][0]);
        __builtin_amdgcn_s_wait_tensorcnt(1);
      } else {
        __builtin_amdgcn_s_wait_tensorcnt(0);
      }
    }
    if (wave == 4) {
      if (t + 1 < 16) {
        tdm_issue_btile(Wbf + (size_t)(16 + t + 1) * (16 * F_), &Bsh[2 + ((t + 1) & 1)][0]);
        __builtin_amdgcn_s_wait_tensorcnt(1);
      } else {
        __builtin_amdgcn_s_wait_tensorcnt(0);
      }
    }
    __syncthreads();   // tile t visible to all waves

    const unsigned short* bBase = &Bsh[stream * 2 + (t & 1)][0] + nrow * AS;

    v8f c = {0.f, 0.f, 0.f, 0.f, 0.f, 0.f, 0.f, 0.f};
#pragma unroll 8
    for (int ks = 0; ks < 32; ++ks) {
      const int k0 = ks * 32;
      union { u32x4 u[2]; v16bf v; } af, bfr;
      af.u[0] = *(const u32x4*)(aBase + k0);        // K = k0 + g*8 .. +7
      af.u[1] = *(const u32x4*)(aBase + k0 + 16);   // K = k0+16 + g*8 .. +7
      const int kk = k0 + g * 8;                    // B: K = k0 + g*8 .. +7 and k0+16 + g*8 .. +7
      const unsigned short* bp = bBase + kk + ((kk >= 512) ? 8 : 0);  // TDM mid-row pad
      bfr.u[0] = *(const u32x4*)bp;
      bfr.u[1] = *(const u32x4*)(bp + 16);
      c = __builtin_amdgcn_wmma_f32_16x16x32_bf16(false, af.v, false, bfr.v,
                                                  (short)0, c, false, false);
    }

    // epilogue: fold tanh(c + bias) * v into per-lane score partials
    const int n = (stream * 16 + t) * 16 + nrow;
    const float bb = sBias[n], vv = sV[n];
#pragma unroll
    for (int r = 0; r < 8; ++r) sp[r] += FAST_TANH(c[r] + bb) * vv;

    __syncthreads();  // everyone done reading buffer before TDM reuses it
  }

  // reduce over the 16 lanes sharing the same row group (sum over n residues)
#pragma unroll
  for (int r = 0; r < 8; ++r) {
    float v = sp[r];
    v += __shfl_xor(v, 1);
    v += __shfl_xor(v, 2);
    v += __shfl_xor(v, 4);
    v += __shfl_xor(v, 8);
    sp[r] = v;
  }
  if ((lane & 15) == 0) {
#pragma unroll
    for (int r = 0; r < 8; ++r)
      atomicAdd(&sScore[msub * 16 + g * 8 + r], sp[r]);   // combine the 2 n-streams
  }
  __syncthreads();
  if (tid < MT) scores[rowBase + tid] = sScore[tid];
}

// ---------------------------------------------------------------------------
// Kernel 2: softmax over L per batch; writes weights output
// ---------------------------------------------------------------------------
__global__ void __launch_bounds__(256)
softmax_kernel(const float* __restrict__ scores, float* __restrict__ wout) {
  __shared__ float red[256];
  const int b = blockIdx.x, tid = threadIdx.x;
  float s[8];
#pragma unroll
  for (int j = 0; j < 8; ++j) s[j] = scores[b * L_ + tid + j * 256];
  float mx = s[0];
#pragma unroll
  for (int j = 1; j < 8; ++j) mx = fmaxf(mx, s[j]);
  red[tid] = mx; __syncthreads();
  for (int o = 128; o > 0; o >>= 1) {
    if (tid < o) red[tid] = fmaxf(red[tid], red[tid + o]);
    __syncthreads();
  }
  mx = red[0]; __syncthreads();
  float sum = 0.f;
#pragma unroll
  for (int j = 0; j < 8; ++j) { s[j] = __expf(s[j] - mx); sum += s[j]; }
  red[tid] = sum; __syncthreads();
  for (int o = 128; o > 0; o >>= 1) {
    if (tid < o) red[tid] += red[tid + o];
    __syncthreads();
  }
  const float inv = 1.f / red[0];
#pragma unroll
  for (int j = 0; j < 8; ++j) wout[b * L_ + tid + j * 256] = s[j] * inv;
}

// ---------------------------------------------------------------------------
// Kernel 3: applied[b,e] = sum_l w[b,l] * emb[b,l,e]  (memory-bound stream)
// ---------------------------------------------------------------------------
__global__ void __launch_bounds__(256)
applied_kernel(const float* __restrict__ w, const float* __restrict__ emb,
               float* __restrict__ applied) {
  __shared__ float wsm[L_];
  const int b = blockIdx.x, tid = threadIdx.x;
  const int e = blockIdx.y * 256 + tid;
  for (int i = tid; i < L_; i += 256) wsm[i] = w[b * L_ + i];
  __syncthreads();
  const float* ep = emb + (size_t)b * L_ * E_ + e;
  float acc = 0.f;
#pragma unroll 4
  for (int l = 0; l < L_; ++l) acc = fmaf(wsm[l], ep[(size_t)l * E_], acc);
  applied[b * E_ + e] = acc;
}

// ---------------------------------------------------------------------------
extern "C" void kernel_launch(void* const* d_in, const int* in_sizes, int n_in,
                              void* d_out, int out_size, void* d_ws, size_t ws_size,
                              hipStream_t stream) {
  const float* emb = (const float*)d_in[0];   // (B,L,E)
  const float* qry = (const float*)d_in[1];   // (B,L,Q)
  const float* W   = (const float*)d_in[2];   // (E, E+Q)
  const float* ba  = (const float*)d_in[3];   // (E,)
  const float* vw  = (const float*)d_in[4];   // (E,)

  float* applied = (float*)d_out;             // (B,1,E)  first
  float* weights = applied + B_ * E_;         // (B,1,L)  second

  unsigned short* Wbf = (unsigned short*)d_ws;                        // 1 MB
  float* scores = (float*)((char*)d_ws + (size_t)E_ * F_ * 2);        // 256 KB

  wprep_kernel<<<512, 256, 0, stream>>>(W, (unsigned*)Wbf);
  attn_scores_kernel<<<(B_ * L_) / MT, 256, 0, stream>>>(qry, emb, Wbf, ba, vw, scores);
  softmax_kernel<<<B_, 256, 0, stream>>>(scores, weights);
  applied_kernel<<<dim3(B_, E_ / 256), 256, 0, stream>>>(weights, emb, applied);
}
